// BlockLocalSelfAttention_72997264163454
// MI455X (gfx1250) — matrix operand order repaired and checked by reading, not compiled, not run on hardware
//
#include <hip/hip_runtime.h>
#include <hip/hip_bf16.h>

typedef __attribute__((ext_vector_type(16))) __bf16 v16bf;
typedef __attribute__((ext_vector_type(8)))  float  v8f;

#define NB_   2
#define NH_   16
#define TT_   4096
#define DD_   64
#define BS_   128
#define NBLK_ 32
#define SLOTS 416          // 384 window keys + 1 global key + 31 masked pad
#define KSTR  72           // sK row stride in halves (16B aligned, bank-spread)
#define VSTR  424          // sVt row stride in halves (16B aligned, bank-spread)
#define NEGF  (-3.402823466e38f)
#define SCALE 0.125f       // 1/sqrt(64)

// ---------------------------------------------------------------------------
// Main kernel: one workgroup per (batch*head, key-block). 8 waves; each wave
// owns a 16-query tile and sweeps 13 chunks of 32 keys with bf16 WMMA.
// ---------------------------------------------------------------------------
__global__ __launch_bounds__(256)
void bla_main(const float* __restrict__ Q, const float* __restrict__ K,
              const float* __restrict__ V, const float* __restrict__ M,
              float* __restrict__ O)
{
    __shared__ __bf16 sK [SLOTS * KSTR];   // keys row-major  [slot][dim]
    __shared__ __bf16 sVt[DD_  * VSTR];    // values transposed [dim][slot]
    __shared__ float  sMask[SLOTS];
    __shared__ float  sPart[256];

    const int    tid  = threadIdx.x;
    const int    blk  = blockIdx.x % NBLK_;
    const int    bh   = blockIdx.x / NBLK_;
    const int    bi   = bh / NH_;
    const size_t base = (size_t)bh * TT_ * DD_;
    const int    p0   = blk * BS_ - BS_;   // window start (may be negative)

    // ---- cooperative stage: K -> sK (bf16), V -> sVt (bf16, transposed) ----
    for (int i = tid; i < SLOTS * (DD_ / 4); i += 256) {
        int slot = i / (DD_ / 4);
        int c4   = (i % (DD_ / 4)) * 4;
        int p    = (slot < 384) ? (p0 + slot) : 0;       // slot 384 = global key 0
        bool ok  = (slot < 384) ? (p >= 0 && p < TT_) : (slot == 384);
        float4 kx = make_float4(0.f, 0.f, 0.f, 0.f);
        float4 vx = make_float4(0.f, 0.f, 0.f, 0.f);
        if (ok) {
            kx = *(const float4*)(K + base + (size_t)p * DD_ + c4);
            vx = *(const float4*)(V + base + (size_t)p * DD_ + c4);
        }
        __bf16* kd = &sK[slot * KSTR + c4];
        kd[0] = (__bf16)kx.x; kd[1] = (__bf16)kx.y;
        kd[2] = (__bf16)kx.z; kd[3] = (__bf16)kx.w;
        sVt[(c4 + 0) * VSTR + slot] = (__bf16)vx.x;
        sVt[(c4 + 1) * VSTR + slot] = (__bf16)vx.y;
        sVt[(c4 + 2) * VSTR + slot] = (__bf16)vx.z;
        sVt[(c4 + 3) * VSTR + slot] = (__bf16)vx.w;
    }
    for (int slot = tid; slot < SLOTS; slot += 256) {
        float m;
        if (slot < 384) {
            int p = p0 + slot;
            // position 0 is masked out of the window (it is the global column)
            m = (p > 0 && p < TT_) ? M[(size_t)bi * TT_ + p] : NEGF;
        } else if (slot == 384) m = 0.0f;   // global column: mask 0
        else                    m = NEGF;   // pad columns
        sMask[slot] = m;
    }
    __syncthreads();

    const int lane = tid & 31, wave = tid >> 5;
    const int lq   = lane & 15, sel = lane >> 4;

    // ---- Q tile for this wave in B-operand layout (lane = query column) ----
    const int    qrow = blk * BS_ + wave * 16 + lq;
    const float* qp   = Q + base + (size_t)qrow * DD_;
    v16bf qb0, qb1;
#pragma unroll
    for (int j = 0; j < 16; ++j) {
        qb0[j] = (__bf16)qp[      sel * 8 + (j & 7) + 16 * (j >> 3)];   // dims  0..31
        qb1[j] = (__bf16)qp[32 +  sel * 8 + (j & 7) + 16 * (j >> 3)];   // dims 32..63
    }

    v8f  o0 = {}, o1 = {}, o2 = {}, o3 = {};  // 16 queries x 64 dims (f32 acc)
    float rsum = 0.0f;                        // per-lane partial softmax denom

    for (int ch = 0; ch < SLOTS / 32; ++ch) {
        const int kb = ch * 32;

        // A tiles of K: 2 key sub-tiles x 2 dim-chunks (A-operand layout)
        v16bf a00, a01, a10, a11;
        {
            const __bf16* r0 = &sK[(kb      + lq) * KSTR];
            const __bf16* r1 = &sK[(kb + 16 + lq) * KSTR];
#pragma unroll
            for (int j = 0; j < 8; ++j) {
                a00[j]     = r0[     8 * sel + j];
                a00[8 + j] = r0[16 + 8 * sel + j];
                a01[j]     = r0[32 + 8 * sel + j];
                a01[8 + j] = r0[48 + 8 * sel + j];
                a10[j]     = r1[     8 * sel + j];
                a10[8 + j] = r1[16 + 8 * sel + j];
                a11[j]     = r1[32 + 8 * sel + j];
                a11[8 + j] = r1[48 + 8 * sel + j];
            }
        }
        // S^T = K . Q^T : lane = query, VGPR = key  (exactly the A layout P needs)
        v8f st0 = {}, st1 = {};
        st0 = __builtin_amdgcn_wmma_f32_16x16x32_bf16(false, a00, false, qb0, (short)0, st0, false, false);
        st0 = __builtin_amdgcn_wmma_f32_16x16x32_bf16(false, a01, false, qb1, (short)0, st0, false, false);
        st1 = __builtin_amdgcn_wmma_f32_16x16x32_bf16(false, a10, false, qb0, (short)0, st1, false, false);
        st1 = __builtin_amdgcn_wmma_f32_16x16x32_bf16(false, a11, false, qb1, (short)0, st1, false, false);

        // P = exp(S*scale + mask); masked columns -> exp(-FLT_MAX) == 0.
        // Conversion straight into the A-operand tile: pure per-lane VALU.
        v16bf ap;
#pragma unroll
        for (int r = 0; r < 8; ++r) {
            float e0 = __expf(st0[r] * SCALE + sMask[kb +      r + 8 * sel]);
            float e1 = __expf(st1[r] * SCALE + sMask[kb + 16 + r + 8 * sel]);
            rsum += e0 + e1;
            ap[r]     = (__bf16)e0;
            ap[8 + r] = (__bf16)e1;
        }

        // O += P . V  (B operand: rows of V^T, contiguous in key per lane)
#pragma unroll
        for (int dt = 0; dt < 4; ++dt) {
            const __bf16* vp2 = &sVt[(dt * 16 + lq) * VSTR + kb + sel * 8];
            v16bf bv;
#pragma unroll
            for (int j = 0; j < 16; ++j) bv[j] = vp2[(j & 7) + 16 * (j >> 3)];
            v8f* op = (dt == 0) ? &o0 : (dt == 1) ? &o1 : (dt == 2) ? &o2 : &o3;
            *op = __builtin_amdgcn_wmma_f32_16x16x32_bf16(false, ap, false, bv, (short)0, *op, false, false);
        }
    }

    // ---- softmax denominator: lanes l and l^16 hold complementary halves ----
    sPart[tid] = rsum;
    __syncthreads();
    float rinv[8];
#pragma unroll
    for (int r = 0; r < 8; ++r) {
        int qq = r + 8 * sel;                     // output row of C-tile VGPR r
        rinv[r] = 1.0f / (sPart[wave * 32 + qq] + sPart[wave * 32 + 16 + qq]);
    }

    // ---- normalize + store (C layout: lane = dim column, VGPR = query row) --
#pragma unroll
    for (int r = 0; r < 8; ++r) {
        int    row    = r + 8 * sel;
        size_t rowoff = base + (size_t)(blk * BS_ + wave * 16 + row) * DD_ + lq;
        O[rowoff +  0] = o0[r] * rinv[r];
        O[rowoff + 16] = o1[r] * rinv[r];
        O[rowoff + 32] = o2[r] * rinv[r];
        O[rowoff + 48] = o3[r] * rinv[r];
    }
}

// ---------------------------------------------------------------------------
// Row-0 special case: full attention over all t keys for query index 0 of
// every (batch, head). 32 rows total -- trivial VALU kernel.
// ---------------------------------------------------------------------------
__global__ __launch_bounds__(256)
void bla_row0(const float* __restrict__ Q, const float* __restrict__ K,
              const float* __restrict__ V, const float* __restrict__ M,
              float* __restrict__ O)
{
    const int    bh   = blockIdx.x;
    const int    bi   = bh / NH_;
    const size_t base = (size_t)bh * TT_ * DD_;
    __shared__ float q0[DD_];
    __shared__ float acc[DD_];
    __shared__ float asum;
    const int tid = threadIdx.x;
    if (tid < DD_) { q0[tid] = Q[base + tid]; acc[tid] = 0.0f; }
    if (tid == 0)  asum = 0.0f;
    __syncthreads();

    float lacc[DD_];
#pragma unroll
    for (int d = 0; d < DD_; ++d) lacc[d] = 0.0f;
    float lsum = 0.0f;

    for (int j = tid; j < TT_; j += 256) {
        const float* kr = K + base + (size_t)j * DD_;
        float s = 0.0f;
#pragma unroll
        for (int d = 0; d < DD_; ++d) s += q0[d] * kr[d];
        float e = __expf(s * SCALE + M[(size_t)bi * TT_ + j]);
        lsum += e;
        const float* vr = V + base + (size_t)j * DD_;
#pragma unroll
        for (int d = 0; d < DD_; ++d) lacc[d] += e * vr[d];
    }
    atomicAdd(&asum, lsum);
    for (int d = 0; d < DD_; ++d) atomicAdd(&acc[d], lacc[d]);
    __syncthreads();
    if (tid < DD_) O[base + tid] = acc[tid] / asum;
}

extern "C" void kernel_launch(void* const* d_in, const int* in_sizes, int n_in,
                              void* d_out, int out_size, void* d_ws, size_t ws_size,
                              hipStream_t stream)
{
    const float* Q = (const float*)d_in[0];
    const float* K = (const float*)d_in[1];
    const float* V = (const float*)d_in[2];
    const float* M = (const float*)d_in[3];
    float*       O = (float*)d_out;

    bla_main<<<NB_ * NH_ * NBLK_, 256, 0, stream>>>(Q, K, V, M, O);
    bla_row0<<<NB_ * NH_,        256, 0, stream>>>(Q, K, V, M, O);
}
